// SLMUMs_45483703665334
// MI455X (gfx1250) — matrix operand order repaired and checked by reading, not compiled, not run on hardware
//
#include <hip/hip_runtime.h>
#include <hip/hip_bf16.h>

// ---------------------------------------------------------------------------
// SLMU spiking-LMU block for MI455X (gfx1250, wave32, WMMA 16x16x32 f16)
// Pipeline: bn1 -> LIF -> (u = x_sp . Wu) -> LIF -> LMU Toeplitz conv (WMMA)
//           -> [x_sp|m] @ Wh^T (WMMA, async double-buffered) -> LIF
//           -> h @ proj^T (WMMA, async double-buffered, transposed store)
//           -> bn2 (stats + apply, in place on d_out)
// ---------------------------------------------------------------------------

typedef __attribute__((ext_vector_type(16))) _Float16 v16h;
typedef __attribute__((ext_vector_type(8)))  _Float16 v8h;
typedef __attribute__((ext_vector_type(8)))  float    v8f;
typedef int v4i_gcc __attribute__((vector_size(16)));   // matches builtin param

#define B_  256
#define C_  512
#define T_  128
#define M_  1024
#define K2_ 1536   // C_ + M_

// ---- CDNA5 async global->LDS copy (ASYNCcnt) with safe fallback ------------
#if defined(__has_builtin)
#if __has_builtin(__builtin_amdgcn_global_load_async_to_lds_b128)
#define HAVE_ASYNC_LDS 1
#endif
#endif

__device__ __forceinline__ void cp16_g2lds(_Float16* lds, const _Float16* g) {
#ifdef HAVE_ASYNC_LDS
    __builtin_amdgcn_global_load_async_to_lds_b128(
        (__attribute__((address_space(1))) v4i_gcc*)(g),
        (__attribute__((address_space(3))) v4i_gcc*)(lds), 0, 0);
#else
    *(v8h*)lds = *(const v8h*)g;
#endif
}
__device__ __forceinline__ void wait_async() {
#ifdef HAVE_ASYNC_LDS
#if __has_builtin(__builtin_amdgcn_s_wait_asynccnt)
    __builtin_amdgcn_s_wait_asynccnt(0);
#else
    asm volatile("s_wait_asynccnt 0x0" ::: "memory");
#endif
#endif
}

// ---- WMMA fragment loaders (contiguous k = 16h + s, same order for A and B) ------
// A 16x32 f16, row-major tile with row stride `ld` halves:
//   lane<16 : M=lane,   K=0..15 contiguous
//   lane>=16: M=lane-16,K=16..31 contiguous
__device__ __forceinline__ v16h ldsA(const _Float16* tile, int ld) {
    int lane = threadIdx.x & 31;
    const _Float16* p = tile + (lane & 15) * ld + ((lane >> 4) << 4);
    v8h lo = *(const v8h*)p;
    v8h hi = *(const v8h*)(p + 8);
    v16h a;
#pragma unroll
    for (int e = 0; e < 8; ++e) { a[e] = lo[e]; a[e + 8] = hi[e]; }
    return a;
}
// B 32x16 f16 from an [N][K] tile, row stride ldk:
//   lane<16 : N=lane,    K=0..15 contiguous
//   lane>=16: N=lane-16, K=16..31 contiguous  -> one 32B load per lane
__device__ __forceinline__ v16h ldB(const _Float16* nk, int ldk) {
    int lane = threadIdx.x & 31;
    return *(const v16h*)(nk + (size_t)(lane & 15) * ldk + ((lane >> 4) << 4));
}

// ---- batchnorm stats over (B,T) per channel; folds g,b into scale/shift ----
__global__ __launch_bounds__(256) void bn_stats_kernel(
    const float* __restrict__ x, const float* __restrict__ g,
    const float* __restrict__ bb, float* __restrict__ scale,
    float* __restrict__ shift) {
    int c = blockIdx.x, tid = threadIdx.x;
    float s = 0.f, s2 = 0.f;
    for (int idx = tid; idx < B_ * T_; idx += 256) {
        int b = idx >> 7, t = idx & (T_ - 1);
        float v = x[((size_t)b * C_ + c) * T_ + t];
        s += v; s2 += v * v;
    }
    __shared__ float rs[256], rq[256];
    rs[tid] = s; rq[tid] = s2; __syncthreads();
    for (int o = 128; o > 0; o >>= 1) {
        if (tid < o) { rs[tid] += rs[tid + o]; rq[tid] += rq[tid + o]; }
        __syncthreads();
    }
    if (tid == 0) {
        float inv = 1.f / (float)(B_ * T_);
        float mu = rs[0] * inv;
        float var = rq[0] * inv - mu * mu;
        float sc = rsqrtf(var + 1e-5f) * g[c];
        scale[c] = sc;
        shift[c] = bb[c] - mu * sc;
    }
}

__global__ void cvt_f16_kernel(const float* __restrict__ src,
                               _Float16* __restrict__ dst, int n) {
    int i = blockIdx.x * 256 + threadIdx.x;
    if (i < n) dst[i] = (_Float16)src[i];
}

// ---- LIF over t on (B,C,T) input; emits spikes f16 in (B,T,C) layout -------
__global__ __launch_bounds__(512) void lif_x_kernel(
    const float* __restrict__ x, const float* __restrict__ scale,
    const float* __restrict__ shift, _Float16* __restrict__ xsp) {
    int b = blockIdx.x, tid = threadIdx.x;           // tid == channel
    __shared__ float tile[C_ * 17];                  // padded, conflict-free
    float v = 0.f;
    float sc = scale[tid], sh = shift[tid];
    for (int tc = 0; tc < T_; tc += 16) {
        for (int i = tid; i < C_ * 16; i += 512) {   // coalesced along t
            int c = i >> 4, tt = i & 15;
            tile[c * 17 + tt] = x[((size_t)b * C_ + c) * T_ + tc + tt];
        }
        __syncthreads();
#pragma unroll
        for (int tt = 0; tt < 16; ++tt) {
            float xn = tile[tid * 17 + tt] * sc + sh;
            v += (xn - v) * 0.5f;                    // TAU = 2
            float s = (v >= 0.5f) ? 1.f : 0.f;       // VTH = 0.5
            v *= (1.f - s);
            xsp[((size_t)b * T_ + tc + tt) * C_ + tid] = (_Float16)s;
        }
        __syncthreads();
    }
}

// ---- u_pre[b,t] = x_sp[b,t,:] . Wu  (one wave per row) ---------------------
__global__ __launch_bounds__(256) void u_pre_kernel(
    const _Float16* __restrict__ xsp, const float* __restrict__ Wu,
    const float* __restrict__ Wub, float* __restrict__ upre) {
    int row = (blockIdx.x * 256 + threadIdx.x) >> 5;
    int lane = threadIdx.x & 31;
    if (row >= B_ * T_) return;
    const _Float16* r = xsp + (size_t)row * C_;
    float acc = 0.f;
#pragma unroll 4
    for (int e = 0; e < 16; ++e) {
        int k = lane * 16 + e;
        acc += (float)r[k] * Wu[k];
    }
#pragma unroll
    for (int m = 16; m >= 1; m >>= 1) acc += __shfl_xor(acc, m, 32);
    if (lane == 0) upre[row] = acc + Wub[0];
}

__global__ void lif_u_kernel(const float* __restrict__ upre,
                             _Float16* __restrict__ usp) {
    int b = blockIdx.x * 256 + threadIdx.x;
    if (b >= B_) return;
    float v = 0.f;
    for (int t = 0; t < T_; ++t) {
        float u = upre[b * T_ + t];
        v += (u - v) * 0.5f;
        float s = (v >= 0.5f) ? 1.f : 0.f;
        v *= (1.f - s);
        usp[b * T_ + t] = (_Float16)s;
    }
}

// ---- LMU conv as Toeplitz matmul: m[b,t,q] = sum_d u[b,t-d] H[q,d] ---------
// Block: (b, t-tile of 16). A = 16x32 Toeplitz chunks built in LDS from the
// spike vector; B = H in [q][d] layout (native). Causality: only nkc chunks.
// kc loops are fully unrolled with a uniform guard so afrag[] indices stay
// compile-time constants (no v_movrels dynamic VGPR indexing).
__global__ __launch_bounds__(256) void lmu_conv_kernel(
    const _Float16* __restrict__ usp, const _Float16* __restrict__ Hh,
    _Float16* __restrict__ m) {
    int b = blockIdx.x;
    int t0 = blockIdx.y * 16;
    int nkc = (blockIdx.y >> 1) + 1;                 // ceil((t0+16)/32) <= 4
    int tid = threadIdx.x, wave = tid >> 5, lane = tid & 31;
    __shared__ _Float16 u_s[T_];
    __shared__ __align__(64) _Float16 a_s[4 * 16 * 32];
    if (tid < T_) u_s[tid] = usp[b * T_ + tid];
    __syncthreads();
    for (int i = tid; i < nkc * 512; i += 256) {     // a_s[kc][tl][dl]
        int tl = (i >> 5) & 15, dl = i & 31, kc = i >> 9;
        int src = t0 + tl - (kc * 32 + dl);
        a_s[i] = (src >= 0) ? u_s[src] : (_Float16)0.f;
    }
    __syncthreads();
    v16h afrag[4];
#pragma unroll
    for (int kc = 0; kc < 4; ++kc)
        if (kc < nkc) afrag[kc] = ldsA(a_s + kc * 512, 32);
#pragma unroll
    for (int j = 0; j < 8; ++j) {                    // 8 q-tiles per wave
        int q0 = (wave + j * 8) * 16;
        v8f acc;
#pragma unroll
        for (int jj = 0; jj < 8; ++jj) acc[jj] = 0.f;
#pragma unroll
        for (int kc = 0; kc < 4; ++kc) {
            if (kc < nkc) {
                v16h bf = ldB(Hh + (size_t)q0 * T_ + kc * 32, T_);
                acc = __builtin_amdgcn_wmma_f32_16x16x32_f16(
                    false, afrag[kc], false, bf, (short)0, acc, false, false);
            }
        }
        int t = t0 + ((lane >> 4) << 3);
        int q = q0 + (lane & 15);
#pragma unroll
        for (int jj = 0; jj < 8; ++jj)
            m[((size_t)b * T_ + t + jj) * M_ + q] = (_Float16)acc[jj];
    }
}

// ---- h_pre = [x_sp | m] (32768x1536) @ Wh^T + bias, f32 out ---------------
// 64x64 block tile, 8 waves (4 row-subtiles x 2 col-pairs).
// A/B tiles staged into double-buffered LDS with async copies; stage(kc+1)
// overlaps the two WMMAs of stage kc; s_wait_asynccnt + barrier per stage.
__global__ __launch_bounds__(256) void gemm_h_kernel(
    const _Float16* __restrict__ xsp, const _Float16* __restrict__ mm,
    const _Float16* __restrict__ Wh, const float* __restrict__ Whb,
    float* __restrict__ hpre) {
    int row0 = blockIdx.x * 64, col0 = blockIdx.y * 64;
    int tid = threadIdx.x, wave = tid >> 5, lane = tid & 31;
    int wr = wave & 3, wc = wave >> 2;
    __shared__ __align__(64) _Float16 a_s[2][64 * 32];
    __shared__ __align__(64) _Float16 b_s[2][64 * 32];
    v8f acc0, acc1;
    {
        float b0 = Whb[col0 + wc * 16 + (lane & 15)];
        float b1 = Whb[col0 + (wc + 2) * 16 + (lane & 15)];
#pragma unroll
        for (int jj = 0; jj < 8; ++jj) { acc0[jj] = b0; acc1[jj] = b1; }
    }
    int rl = tid >> 2, koff = (tid & 3) * 8;
    auto stage = [&](int kc, int buf) {
        int k = kc * 32;
        const _Float16* asrc = (k < C_)
            ? xsp + (size_t)(row0 + rl) * C_ + k + koff
            : mm  + (size_t)(row0 + rl) * M_ + (k - C_) + koff;
        cp16_g2lds(a_s[buf] + rl * 32 + koff, asrc);
        cp16_g2lds(b_s[buf] + rl * 32 + koff,
                   Wh + (size_t)(col0 + rl) * K2_ + k + koff);
    };
    const int NK = K2_ / 32;
    stage(0, 0);
    wait_async();
    __syncthreads();
    for (int kc = 0; kc < NK; ++kc) {
        int cur = kc & 1;
        if (kc + 1 < NK) stage(kc + 1, cur ^ 1);     // overlap with WMMA
        v16h a  = ldsA(a_s[cur] + wr * 512, 32);
        v16h b0 = ldB(b_s[cur] + wc * 512, 32);
        v16h b1 = ldB(b_s[cur] + (wc + 2) * 512, 32);
        acc0 = __builtin_amdgcn_wmma_f32_16x16x32_f16(
            false, a, false, b0, (short)0, acc0, false, false);
        acc1 = __builtin_amdgcn_wmma_f32_16x16x32_f16(
            false, a, false, b1, (short)0, acc1, false, false);
        wait_async();
        __syncthreads();
    }
    int r  = row0 + wr * 16 + ((lane >> 4) << 3);
    int c0 = col0 + wc * 16 + (lane & 15);
#pragma unroll
    for (int jj = 0; jj < 8; ++jj) {
        hpre[(size_t)(r + jj) * C_ + c0]      = acc0[jj];
        hpre[(size_t)(r + jj) * C_ + c0 + 32] = acc1[jj];
    }
}

// ---- LIF over t on (B,T,C) f32 input -> spikes f16 (B,T,C) ----------------
__global__ __launch_bounds__(512) void lif_h_kernel(
    const float* __restrict__ hpre, _Float16* __restrict__ hsp) {
    int b = blockIdx.x, c = threadIdx.x;
    float v = 0.f;
    for (int t = 0; t < T_; ++t) {
        float x = hpre[((size_t)b * T_ + t) * C_ + c];   // coalesced
        v += (x - v) * 0.5f;
        float s = (v >= 0.5f) ? 1.f : 0.f;
        v *= (1.f - s);
        hsp[((size_t)b * T_ + t) * C_ + c] = (_Float16)s;
    }
}

// ---- y[b,o,t] = h (32768x512) @ proj^T + bias; LDS-transposed store -------
__global__ __launch_bounds__(256) void gemm_y_kernel(
    const _Float16* __restrict__ hsp, const _Float16* __restrict__ Pw,
    const float* __restrict__ Pb, float* __restrict__ y) {
    int row0 = blockIdx.x * 64, col0 = blockIdx.y * 64;
    int tid = threadIdx.x, wave = tid >> 5, lane = tid & 31;
    int wr = wave & 3, wc = wave >> 2;
    __shared__ __align__(64) _Float16 a_s[2][64 * 32];
    __shared__ __align__(64) _Float16 b_s[2][64 * 32];
    __shared__ float o_s[64 * 65];                   // padded transpose buf
    v8f acc0, acc1;
    {
        float b0 = Pb[col0 + wc * 16 + (lane & 15)];
        float b1 = Pb[col0 + (wc + 2) * 16 + (lane & 15)];
#pragma unroll
        for (int jj = 0; jj < 8; ++jj) { acc0[jj] = b0; acc1[jj] = b1; }
    }
    int rl = tid >> 2, koff = (tid & 3) * 8;
    auto stage = [&](int kc, int buf) {
        int k = kc * 32;
        cp16_g2lds(a_s[buf] + rl * 32 + koff,
                   hsp + (size_t)(row0 + rl) * C_ + k + koff);
        cp16_g2lds(b_s[buf] + rl * 32 + koff,
                   Pw + (size_t)(col0 + rl) * C_ + k + koff);
    };
    const int NK = C_ / 32;
    stage(0, 0);
    wait_async();
    __syncthreads();
    for (int kc = 0; kc < NK; ++kc) {
        int cur = kc & 1;
        if (kc + 1 < NK) stage(kc + 1, cur ^ 1);
        v16h a  = ldsA(a_s[cur] + wr * 512, 32);
        v16h b0 = ldB(b_s[cur] + wc * 512, 32);
        v16h b1 = ldB(b_s[cur] + (wc + 2) * 512, 32);
        acc0 = __builtin_amdgcn_wmma_f32_16x16x32_f16(
            false, a, false, b0, (short)0, acc0, false, false);
        acc1 = __builtin_amdgcn_wmma_f32_16x16x32_f16(
            false, a, false, b1, (short)0, acc1, false, false);
        wait_async();
        __syncthreads();
    }
    int tl  = wr * 16 + ((lane >> 4) << 3);
    int ol0 = wc * 16 + (lane & 15);
#pragma unroll
    for (int jj = 0; jj < 8; ++jj) {
        o_s[(ol0)      * 65 + tl + jj] = acc0[jj];
        o_s[(ol0 + 32) * 65 + tl + jj] = acc1[jj];
    }
    __syncthreads();
    int b  = row0 >> 7;                              // T_ == 128
    int t0 = row0 & 127;
    int ol = tid >> 2, tof = (tid & 3) * 16;         // 16 floats per thread
    float4* dst = (float4*)(y + ((size_t)b * C_ + col0 + ol) * T_ + t0 + tof);
    const float* sp = o_s + ol * 65 + tof;
#pragma unroll
    for (int q = 0; q < 4; ++q)
        dst[q] = make_float4(sp[4*q], sp[4*q+1], sp[4*q+2], sp[4*q+3]);
}

__global__ void bn2_apply_kernel(float* __restrict__ y,
                                 const float* __restrict__ scale,
                                 const float* __restrict__ shift) {
    size_t i = ((size_t)blockIdx.x * 256 + threadIdx.x) * 4;
    int c = (int)((i >> 7) & (C_ - 1));
    float4* p = (float4*)(y + i);
    float4 v = *p;
    float sc = scale[c], sh = shift[c];
    v.x = v.x * sc + sh; v.y = v.y * sc + sh;
    v.z = v.z * sc + sh; v.w = v.w * sc + sh;
    *p = v;
}

// ---------------------------------------------------------------------------
extern "C" void kernel_launch(void* const* d_in, const int* in_sizes, int n_in,
                              void* d_out, int out_size, void* d_ws,
                              size_t ws_size, hipStream_t stream) {
    (void)in_sizes; (void)n_in; (void)out_size; (void)ws_size;
    const float* x     = (const float*)d_in[0];
    const float* bn1_g = (const float*)d_in[1];
    const float* bn1_b = (const float*)d_in[2];
    const float* Wu_w  = (const float*)d_in[3];
    const float* Wu_b  = (const float*)d_in[4];
    const float* Wh_w  = (const float*)d_in[5];
    const float* Wh_b  = (const float*)d_in[6];
    const float* Pw    = (const float*)d_in[7];
    const float* Pb    = (const float*)d_in[8];
    const float* bn2_g = (const float*)d_in[9];
    const float* bn2_b = (const float*)d_in[10];
    const float* H     = (const float*)d_in[11];
    float* out = (float*)d_out;
    char* ws = (char*)d_ws;
    const size_t MB = 1024ull * 1024ull;

    float*    sc1  = (float*)(ws + 0);
    float*    sh1  = (float*)(ws + 4096);
    float*    sc2  = (float*)(ws + 8192);
    float*    sh2  = (float*)(ws + 12288);
    float*    upre = (float*)(ws + 16384);                       // 128 KB
    _Float16* usp  = (_Float16*)(ws + 16384 + 131072);           // 64 KB
    _Float16* Hh   = (_Float16*)(ws + 16384 + 131072 + 65536);   // 256 KB
    _Float16* Whh  = (_Float16*)(ws + 16384 + 131072 + 65536 + 262144); // 1.5MB
    _Float16* Pwh  = (_Float16*)(ws + 16384 + 131072 + 65536 + 262144 + 1572864);
    _Float16* xsp  = (_Float16*)(ws + 4   * MB);                 // 32 MB
    _Float16* mbuf = (_Float16*)(ws + 36  * MB);                 // 64 MB
    float*    hpre = (float*)(ws + 100 * MB);                    // 64 MB
    _Float16* hsp  = (_Float16*)(ws + 164 * MB);                 // 32 MB

    bn_stats_kernel<<<C_, 256, 0, stream>>>(x, bn1_g, bn1_b, sc1, sh1);
    cvt_f16_kernel<<<(M_ * T_ + 255) / 256, 256, 0, stream>>>(H, Hh, M_ * T_);
    cvt_f16_kernel<<<(C_ * K2_ + 255) / 256, 256, 0, stream>>>(Wh_w, Whh, C_ * K2_);
    cvt_f16_kernel<<<(C_ * C_ + 255) / 256, 256, 0, stream>>>(Pw, Pwh, C_ * C_);
    lif_x_kernel<<<B_, 512, 0, stream>>>(x, sc1, sh1, xsp);
    u_pre_kernel<<<(B_ * T_) / 8, 256, 0, stream>>>(xsp, Wu_w, Wu_b, upre);
    lif_u_kernel<<<1, 256, 0, stream>>>(upre, usp);
    lmu_conv_kernel<<<dim3(B_, T_ / 16), 256, 0, stream>>>(usp, Hh, mbuf);
    gemm_h_kernel<<<dim3((B_ * T_) / 64, C_ / 64), 256, 0, stream>>>(
        xsp, mbuf, Whh, Wh_b, hpre);
    lif_h_kernel<<<B_, 512, 0, stream>>>(hpre, hsp);
    gemm_y_kernel<<<dim3((B_ * T_) / 64, C_ / 64), 256, 0, stream>>>(
        hsp, Pwh, Pb, out);
    bn_stats_kernel<<<C_, 256, 0, stream>>>(out, bn2_g, bn2_b, sc2, sh2);
    bn2_apply_kernel<<<(B_ * C_ * T_ / 4) / 256, 256, 0, stream>>>(out, sc2, sh2);
}
